// MambaBlock_85315230367965
// MI455X (gfx1250) — compile-verified, matrix operand order checked by reading, not run on hardware
//
#include <hip/hip_runtime.h>
#include <hip/hip_bf16.h>
#include <stdint.h>

#define T_SEQ   512
#define NEMBD   1024
#define DINNER  2048
#define DSTATE  64
#define NHEADS  32
#define HEADDIM 64
#define DCONV   4
#define PROJ    4256   // 2*DINNER + NHEADS + 2*DSTATE
#define EPSV    1e-6f

#define BK  32          // K-step (one WMMA K per step)
#define BNT 512         // block N tile = 8 waves * 64
#define WN  64          // per-wave N tile (4 x 16-col subtiles)

typedef float  f32x8   __attribute__((ext_vector_type(8)));
typedef __bf16 bf16x8  __attribute__((ext_vector_type(8)));
typedef __bf16 bf16x16 __attribute__((ext_vector_type(16)));
typedef unsigned int u32x4 __attribute__((ext_vector_type(4)));
typedef int i32x4 __attribute__((ext_vector_type(4)));
typedef int i32x8 __attribute__((ext_vector_type(8)));

#if defined(__has_builtin)
#  if __has_builtin(__builtin_amdgcn_tensor_load_to_lds)
#    define HAVE_TDM 1
#  endif
#endif
#ifndef HAVE_TDM
#  define HAVE_TDM 0
#endif

__device__ __forceinline__ unsigned short f32_to_bf16(float f) {
    union { float f; unsigned int u; } x; x.f = f;
    unsigned int u = x.u;
    unsigned int r = u + 0x7FFFu + ((u >> 16) & 1u);   // round-to-nearest-even
    return (unsigned short)(r >> 16);
}

__device__ __forceinline__ float siluf(float x) { return x / (1.0f + __expf(-x)); }
__device__ __forceinline__ float softplusf(float x) { return (x > 20.0f) ? x : log1pf(__expf(x)); }

// Build a WMMA bf16 fragment (A 16x32 or B 32x16 layout) from two aligned 16B chunks.
__device__ __forceinline__ bf16x16 load_bf16x16(const unsigned short* p0, const unsigned short* p1) {
    bf16x8 lo = *reinterpret_cast<const bf16x8*>(p0);
    bf16x8 hi = *reinterpret_cast<const bf16x8*>(p1);
    bf16x16 r;
#pragma unroll
    for (int i = 0; i < 8; ++i) { r[i] = lo[i]; r[i + 8] = hi[i]; }
    return r;
}

#if HAVE_TDM
// Issue a 2D TDM load: tile_n rows of tile_k contiguous bf16 elements,
// row pitch = stride_elems, into LDS at byte offset lds_off.
// D# layout per CDNA5 ISA.
__device__ __forceinline__ void tdm_load_2d_bf16(unsigned lds_off, const unsigned short* gptr,
                                                 unsigned tile_k, unsigned tile_n,
                                                 unsigned tensor_k, unsigned tensor_n,
                                                 unsigned long long stride_elems) {
    unsigned long long ga = (unsigned long long)(uintptr_t)gptr;
    u32x4 g0;
    g0[0] = 1u;                                                 // count=1, user mode
    g0[1] = lds_off;                                            // lds_addr (bytes)
    g0[2] = (unsigned)(ga & 0xFFFFFFFFu);                       // global_addr[31:0]
    g0[3] = (unsigned)((ga >> 32) & 0x1FFFFFFu) | (2u << 30);   // addr[56:32] | type=2
    i32x8 g1;
    g1[0] = (int)(1u << 16);                                    // data_size=1 (2B); mask=0
    g1[1] = (int)((tensor_k & 0xFFFFu) << 16);                  // tensor_dim0[15:0]
    g1[2] = (int)(((tensor_k >> 16) & 0xFFFFu) | ((tensor_n & 0xFFFFu) << 16));
    g1[3] = (int)(((tensor_n >> 16) & 0xFFFFu) | ((tile_k & 0xFFFFu) << 16));
    g1[4] = (int)(tile_n & 0xFFFFu);                            // tile_dim1 (tile_dim2=0)
    g1[5] = (int)(unsigned)(stride_elems & 0xFFFFFFFFu);        // dim0_stride[31:0]
    g1[6] = (int)(unsigned)((stride_elems >> 32) & 0xFFFFu);    // dim0_stride[47:32]
    g1[7] = 0;
    i32x4 z4 = {0, 0, 0, 0};
#if __has_include(<hip/amd_detail/amd_gfx1250_TDM.h>)
    i32x8 z8 = {0, 0, 0, 0, 0, 0, 0, 0};
    __builtin_amdgcn_tensor_load_to_lds(g0, g1, z4, z4, z8, 0);
#else
    __builtin_amdgcn_tensor_load_to_lds(g0, g1, z4, z4, 0);
#endif
}
#endif

// ---------------------------------------------------------------------------
// RMSNorm over E=1024 + cast to bf16 (GEMM1 A operand)
// ---------------------------------------------------------------------------
__global__ __launch_bounds__(256) void k_rmsnorm_cast(const float* __restrict__ x,
                                                      const float* __restrict__ scale,
                                                      unsigned short* __restrict__ xn) {
    __shared__ float red[256];
    const int t = blockIdx.x;
    const float* row = x + (size_t)t * NEMBD;
    float s = 0.f;
#pragma unroll
    for (int i = 0; i < NEMBD / 256; ++i) {
        float v = row[threadIdx.x + i * 256];
        s += v * v;
    }
    red[threadIdx.x] = s;
    __syncthreads();
    for (int off = 128; off > 0; off >>= 1) {
        if (threadIdx.x < off) red[threadIdx.x] += red[threadIdx.x + off];
        __syncthreads();
    }
    const float inv = rsqrtf(red[0] * (1.0f / (float)NEMBD) + EPSV);
    unsigned short* drow = xn + (size_t)t * NEMBD;
#pragma unroll
    for (int i = 0; i < NEMBD / 256; ++i) {
        int e = threadIdx.x + i * 256;
        drow[e] = f32_to_bf16(row[e] * inv * scale[e]);
    }
}

// ---------------------------------------------------------------------------
// fp32 [K][N] -> bf16 transposed [N][K], 32x32 LDS-tiled (one-time weight prep)
// ---------------------------------------------------------------------------
__global__ __launch_bounds__(256) void k_cast_transpose_bf16(const float* __restrict__ src,
                                                             unsigned short* __restrict__ dst,
                                                             int K, int N) {
    __shared__ unsigned short tile[32][33];
    const int kt = blockIdx.y * 32;
    const int nt = blockIdx.x * 32;
    const int tx = threadIdx.x & 31;
    const int ty = threadIdx.x >> 5;     // 0..7
#pragma unroll
    for (int i = 0; i < 32; i += 8) {
        int k = kt + ty + i, n = nt + tx;
        float v = (k < K && n < N) ? src[(size_t)k * N + n] : 0.f;
        tile[ty + i][tx] = f32_to_bf16(v);
    }
    __syncthreads();
#pragma unroll
    for (int i = 0; i < 32; i += 8) {
        int n = nt + ty + i, k = kt + tx;
        if (n < N && k < K) dst[(size_t)n * K + k] = tile[tx][ty + i];
    }
}

// ---------------------------------------------------------------------------
// bf16 WMMA GEMM: C[M,N] (+residual) = A[M,K] @ Bt^T, Bt is [N][K] (K-major).
// Block: 256 threads = 8 waves; block tile 16M x 512N; per-wave 16x64 (4 WMMA).
// B tiles (512 x 32 bf16 = 32 KB) staged into LDS by the Tensor Data Mover,
// double-buffered: s_wait_tensorcnt (all waves; non-issuers pass trivially)
// + workgroup barrier handoff.
// ---------------------------------------------------------------------------
__global__ __launch_bounds__(256) void k_gemm_bf16_wmma(
    const unsigned short* __restrict__ A, const unsigned short* __restrict__ Bt,
    float* __restrict__ C, const float* __restrict__ residual,
    int M, int N, int K) {
    __shared__ __align__(16) unsigned short ldsB[2][BNT * BK];   // [buf][n][k], 2 x 32 KB

    const int lane = threadIdx.x & 31;
    const int wave = threadIdx.x >> 5;
    const int m0   = blockIdx.y * 16;
    const int n0   = blockIdx.x * BNT;

    // A fragment layout (ISA 7.12.2, 16-bit 16x32): row = lane&15, kbase = (lane>>4)*8
    const int rowA = m0 + (lane & 15);
    const int kb   = (lane >> 4) << 3;
    const int colL = lane & 15;

    const unsigned tn = (unsigned)((N > n0) ? (N - n0) : 0);   // rows left -> TDM zero-fills OOB
#if HAVE_TDM
    const unsigned lds_base = (unsigned)(uintptr_t)(&ldsB[0][0]);
#endif

    f32x8 acc[4];
#pragma unroll
    for (int s = 0; s < 4; ++s) acc[s] = f32x8{};

#if HAVE_TDM
    if (wave == 0)
        tdm_load_2d_bf16(lds_base, Bt + (size_t)n0 * K, BK, BNT, (unsigned)K, tn,
                         (unsigned long long)K);
#endif

    int cur = 0;
    for (int k0 = 0; k0 < K; k0 += BK) {
        const bool more = (k0 + BK) < K;
#if HAVE_TDM
        if (more) {
            __syncthreads();   // buf[cur^1] consumers from previous iteration are done
            if (wave == 0)
                tdm_load_2d_bf16(lds_base + (unsigned)(cur ^ 1) * (BNT * BK * 2),
                                 Bt + (size_t)n0 * K + (k0 + BK), BK, BNT,
                                 (unsigned)K, tn, (unsigned long long)K);
            __builtin_amdgcn_s_wait_tensorcnt(1);   // uniform: TENSORcnt==0 waves pass
        } else {
            __builtin_amdgcn_s_wait_tensorcnt(0);
        }
        __syncthreads();
#else
        // Fallback: cooperative contiguous copy (layout in LDS == layout in Bt)
        __syncthreads();
        {
            unsigned short* dstb = &ldsB[cur][0];
#pragma unroll
            for (int p = 0; p < 8; ++p) {
                int piece = threadIdx.x + p * 256;     // 0..2047: (row n, 16B chunk c)
                int n = piece >> 2;
                int c = (piece & 3) << 3;
                int row = n0 + n; if (row >= N) row = N - 1;
                *reinterpret_cast<uint4*>(dstb + n * BK + c) =
                    *reinterpret_cast<const uint4*>(Bt + (size_t)row * K + k0 + c);
            }
        }
        __syncthreads();
#endif
        const unsigned short* pa = A + (size_t)rowA * K + k0 + kb;
        bf16x16 afrag = load_bf16x16(pa, pa + 16);

        // Load all four B fragments into independent registers first so the
        // scheduler can overlap DS returns with WMMA issue (partial dscnt waits).
        const unsigned short* lb = &ldsB[cur][0];
        bf16x16 bfrag[4];
#pragma unroll
        for (int s = 0; s < 4; ++s) {
            const unsigned short* pb = lb + (wave * WN + s * 16 + colL) * BK + kb;
            bfrag[s] = load_bf16x16(pb, pb + 16);
        }
#pragma unroll
        for (int s = 0; s < 4; ++s) {
            acc[s] = __builtin_amdgcn_wmma_f32_16x16x32_bf16(false, afrag, false, bfrag[s],
                                                             (short)0, acc[s], false, false);
        }
        cur ^= 1;
    }

    // C/D layout: acc elem r -> row m0 + (lane>>4)*8 + r, col = subtile + (lane&15)
    const int rbase = m0 + ((lane >> 4) << 3);
#pragma unroll
    for (int s = 0; s < 4; ++s) {
        int col = n0 + wave * WN + s * 16 + colL;
        if (col < N) {
#pragma unroll
            for (int r = 0; r < 8; ++r) {
                size_t o = (size_t)(rbase + r) * N + col;
                float v = acc[s][r];
                if (residual) v += residual[o];
                C[o] = v;
            }
        }
    }
}

// ---------------------------------------------------------------------------
// dt = softplus(dt_raw + dt_bias); A_bar = exp(-exp(A_log) * dt)
// ---------------------------------------------------------------------------
__global__ void k_dt(const float* __restrict__ proj, const float* __restrict__ A_log,
                     const float* __restrict__ dt_bias,
                     float* __restrict__ dt, float* __restrict__ abar) {
    const int t = blockIdx.x;
    const int hh = threadIdx.x;
    float draw = proj[(size_t)t * PROJ + 2 * DINNER + hh];
    float d = softplusf(draw + dt_bias[hh]);
    float a = -__expf(A_log[hh]);
    dt[t * NHEADS + hh]   = d;
    abar[t * NHEADS + hh] = __expf(a * d);
}

// ---------------------------------------------------------------------------
// Depthwise causal conv (4 taps) + SiLU over the ssm_in slice of proj
// ---------------------------------------------------------------------------
__global__ void k_conv_silu(const float* __restrict__ proj, const float* __restrict__ cw,
                            const float* __restrict__ cb, float* __restrict__ h) {
    int idx = blockIdx.x * blockDim.x + threadIdx.x;
    if (idx >= T_SEQ * DINNER) return;
    int t = idx / DINNER;
    int c = idx - t * DINNER;
    float acc = cb[c];
#pragma unroll
    for (int k = 0; k < DCONV; ++k) {
        int tt = t + k - (DCONV - 1);
        if (tt >= 0) acc += cw[k * DINNER + c] * proj[(size_t)tt * PROJ + DINNER + c];
    }
    h[idx] = siluf(acc);
}

// ---------------------------------------------------------------------------
// Sequential SSM scan: one block per head; each thread owns 16 of the 64x64
// state in registers. y is SiLU-gated and written as bf16 (GEMM2 A operand).
// ---------------------------------------------------------------------------
__global__ __launch_bounds__(256) void k_scan(const float* __restrict__ proj,
                                              const float* __restrict__ h,
                                              const float* __restrict__ dt,
                                              const float* __restrict__ abar,
                                              unsigned short* __restrict__ yb) {
    const int head = blockIdx.x;
    const int tid  = threadIdx.x;
    const int p    = tid >> 2;          // head-dim index 0..63
    const int nb   = (tid & 3) << 4;    // 16-wide state slice

    float s[16];
#pragma unroll
    for (int j = 0; j < 16; ++j) s[j] = 0.f;

    for (int t = 0; t < T_SEQ; ++t) {
        const float* prow = proj + (size_t)t * PROJ;
        float dtv = dt[t * NHEADS + head];
        float ab  = abar[t * NHEADS + head];
        float xv  = h[(size_t)t * DINNER + head * HEADDIM + p];
        float coef = dtv * xv;

        const float4* Bp = reinterpret_cast<const float4*>(prow + 2 * DINNER + NHEADS + nb);
        const float4* Cp = reinterpret_cast<const float4*>(prow + 2 * DINNER + NHEADS + DSTATE + nb);

        float partial = 0.f;
#pragma unroll
        for (int q = 0; q < 4; ++q) {
            float4 bv = Bp[q];
            float4 cv = Cp[q];
            float* sb = &s[q * 4];
            sb[0] = ab * sb[0] + coef * bv.x; partial += sb[0] * cv.x;
            sb[1] = ab * sb[1] + coef * bv.y; partial += sb[1] * cv.y;
            sb[2] = ab * sb[2] + coef * bv.z; partial += sb[2] * cv.z;
            sb[3] = ab * sb[3] + coef * bv.w; partial += sb[3] * cv.w;
        }
        partial += __shfl_xor(partial, 1, 32);
        partial += __shfl_xor(partial, 2, 32);

        if ((tid & 3) == 0) {
            float g = prow[head * HEADDIM + p];   // gate slice
            yb[(size_t)t * DINNER + head * HEADDIM + p] = f32_to_bf16(partial * siluf(g));
        }
    }
}

// ---------------------------------------------------------------------------
extern "C" void kernel_launch(void* const* d_in, const int* in_sizes, int n_in,
                              void* d_out, int out_size, void* d_ws, size_t ws_size,
                              hipStream_t stream) {
    (void)in_sizes; (void)n_in; (void)out_size; (void)ws_size;
    const float* x          = (const float*)d_in[0];
    const float* norm_scale = (const float*)d_in[1];
    const float* w_in       = (const float*)d_in[2];
    const float* conv_w     = (const float*)d_in[3];
    const float* conv_b     = (const float*)d_in[4];
    const float* A_log      = (const float*)d_in[5];
    const float* dt_bias    = (const float*)d_in[6];
    const float* w_out      = (const float*)d_in[7];
    float* out = (float*)d_out;

    char* ws = (char*)d_ws;
    size_t off = 0;
    auto alloc = [&](size_t bytes) { void* pp = ws + off; off += (bytes + 255) & ~(size_t)255; return pp; };
    unsigned short* xn    = (unsigned short*)alloc((size_t)T_SEQ * NEMBD * 2);
    unsigned short* winT  = (unsigned short*)alloc((size_t)NEMBD * PROJ * 2);    // [PROJ][NEMBD]
    unsigned short* woutT = (unsigned short*)alloc((size_t)DINNER * NEMBD * 2);  // [NEMBD][DINNER]
    float* proj = (float*)alloc((size_t)T_SEQ * PROJ * 4);
    float* hbuf = (float*)alloc((size_t)T_SEQ * DINNER * 4);
    float* dtb  = (float*)alloc((size_t)T_SEQ * NHEADS * 4);
    float* ab   = (float*)alloc((size_t)T_SEQ * NHEADS * 4);
    unsigned short* yb = (unsigned short*)alloc((size_t)T_SEQ * DINNER * 2);

    k_rmsnorm_cast<<<T_SEQ, 256, 0, stream>>>(x, norm_scale, xn);

    // One-time weight prep: cast + transpose to K-major bf16
    {
        dim3 gt1((PROJ + 31) / 32, NEMBD / 32);
        k_cast_transpose_bf16<<<gt1, 256, 0, stream>>>(w_in, winT, NEMBD, PROJ);
        dim3 gt2((NEMBD + 31) / 32, DINNER / 32);
        k_cast_transpose_bf16<<<gt2, 256, 0, stream>>>(w_out, woutT, DINNER, NEMBD);
    }

    dim3 g1((PROJ + BNT - 1) / BNT, T_SEQ / 16);
    k_gemm_bf16_wmma<<<g1, 256, 0, stream>>>(xn, winT, proj, nullptr, T_SEQ, PROJ, NEMBD);

    k_dt<<<T_SEQ, NHEADS, 0, stream>>>(proj, A_log, dt_bias, dtb, ab);
    k_conv_silu<<<(T_SEQ * DINNER + 255) / 256, 256, 0, stream>>>(proj, conv_w, conv_b, hbuf);
    k_scan<<<NHEADS, 256, 0, stream>>>(proj, hbuf, dtb, ab, yb);

    dim3 g2((NEMBD + BNT - 1) / BNT, T_SEQ / 16);
    k_gemm_bf16_wmma<<<g2, 256, 0, stream>>>(yb, woutT, out, x, T_SEQ, NEMBD, DINNER);
}
